// MelPCENPreprocessor_55722905698373
// MI455X (gfx1250) — compile-verified, matrix operand order checked by reading, not run on hardware
//
#include <hip/hip_runtime.h>
#include <hip/hip_bf16.h>
#include <math.h>

typedef __attribute__((ext_vector_type(16))) _Float16 v16h;
typedef __attribute__((ext_vector_type(8)))  float    v8f;

// ---------------- constants ----------------
#define BATCH   2048
#define NSAMP   32000
#define NFFT    400
#define HOP     160
#define PADW    200
#define TFRAMES 201      // (32400-400)/160+1
#define TPAD    208      // 13 tiles of 16
#define NMELS   128
#define NFREQ   201
#define K1      416      // NFFT padded to mult of 32 (13 k-chunks)
#define FPAD    224      // NFREQ padded to mult of 32 (14 n-chunks / 7 k-chunks)
#define NWAVES  13       // one wave per 16-frame M tile
#define BLOCKT  (NWAVES*32)

#define NB_ELE  (13*14*32*16)   // swizzled Bre/Bim element count = 93184
#define FB_ELE  (7*8*32*16)     // swizzled melFB element count   = 28672

// k-offset within a 32-wide K chunk for f16 A/B fragments (ISA 7.12.2):
// lane half (0: lanes 0-15, 1: lanes 16-31), element e = 2*vgpr+pair
__device__ __forceinline__ int kmap(int e, int half) {
  return ((e & 8) << 1) | (half << 3) | (e & 7);
}

__device__ __forceinline__ v8f wmma16(v16h a, v16h b, v8f c) {
  return __builtin_amdgcn_wmma_f32_16x16x32_f16(false, a, false, b, (short)0, c, false, false);
}

// ---------------- setup: build swizzled DFT + mel matrices ----------------
__global__ void mel_setup_kernel(_Float16* __restrict__ BreSw,
                                 _Float16* __restrict__ BimSw,
                                 _Float16* __restrict__ FBSw) {
  int gid = blockIdx.x * blockDim.x + threadIdx.x;
  if (gid < 2 * NB_ELE) {
    int which = gid / NB_ELE;
    int r = gid % NB_ELE;                 // r = ((kc*14+nc)*32+lane)*16+e
    int e    = r & 15;
    int lane = (r >> 4) & 31;
    int q    = r >> 9;                    // kc*14+nc
    int nc   = q % 14;
    int kc   = q / 14;
    int k = kc * 32 + kmap(e, lane >> 4);
    int f = nc * 16 + (lane & 15);
    float val = 0.0f;
    if (k < NFFT && f < NFREQ) {
      float w = 0.5f * (1.0f - cosf(6.2831853071795864f * (float)k / (float)NFFT));
      int mrem = (k * f) % NFFT;          // exact phase reduction
      float ang = 6.2831853071795864f * (float)mrem / (float)NFFT;
      val = (which == 0) ? (w * cosf(ang)) : (-w * sinf(ang));
    }
    ((which == 0) ? BreSw : BimSw)[r] = (_Float16)val;
  } else {
    int r = gid - 2 * NB_ELE;
    if (r < FB_ELE) {                     // r = ((kc2*8+n2)*32+lane)*16+e
      int e    = r & 15;
      int lane = (r >> 4) & 31;
      int q    = r >> 9;                  // kc2*8+n2
      int n2   = q & 7;
      int kc2  = q >> 3;
      int k   = kc2 * 32 + kmap(e, lane >> 4);   // freq bin (row of FB)
      int mel = n2 * 16 + (lane & 15);           // mel col
      float val = 0.0f;
      if (k < NFREQ) {
        float freq   = (float)k * 40.0f;         // (SR/2)/200
        float melmax = 2595.0f * log10f(1.0f + 8000.0f / 700.0f);
        float step   = melmax / 129.0f;          // N_MELS+2 points
        float fp0 = 700.0f * (exp10f((step * (float)(mel    )) / 2595.0f) - 1.0f);
        float fp1 = 700.0f * (exp10f((step * (float)(mel + 1)) / 2595.0f) - 1.0f);
        float fp2 = 700.0f * (exp10f((step * (float)(mel + 2)) / 2595.0f) - 1.0f);
        float dn = (freq - fp0) / (fp1 - fp0);
        float up = (fp2 - freq) / (fp2 - fp1);
        val = fmaxf(0.0f, fminf(dn, up));
      }
      FBSw[r] = (_Float16)val;
    }
  }
}

// ---------------- fused mel + PCEN + resize, one workgroup per batch row ----------------
__global__ __launch_bounds__(BLOCKT)
void mel_pcen_fused_kernel(const float* __restrict__ audio,
                           const _Float16* __restrict__ BreSw,
                           const _Float16* __restrict__ BimSw,
                           const _Float16* __restrict__ FBSw,
                           float* __restrict__ out) {
  // power staged in A-fragment lane order for matmul2: [wave][kc2][lane][e]
  __shared__ alignas(32) _Float16 powLDS[NWAVES][7][32][16];   // 93,184 B
  __shared__ float melLDS[TFRAMES][NMELS];                     // 102,912 B

  const int b    = blockIdx.x;
  const int tid  = threadIdx.x;
  const int wave = tid >> 5;      // == M tile index (0..12)
  const int lane = tid & 31;
  const int lm   = lane & 15;
  const int lh   = lane >> 4;

  const float* arow = audio + (size_t)b * NSAMP;

  // ---- load all 13 A fragments (16 frames x K=416 raw samples, reflect pad) ----
  int t = wave * 16 + lm;
  int tc = t > (TFRAMES - 1) ? (TFRAMES - 1) : t;   // clamp pad rows (results discarded)
  v16h aF[13];
  #pragma unroll
  for (int kc = 0; kc < 13; ++kc) {
    v16h a;
    #pragma unroll
    for (int e = 0; e < 16; ++e) {
      int k = kc * 32 + kmap(e, lh);
      float v = 0.0f;
      if (k < NFFT) {
        int s = tc * HOP + k - PADW;
        s = (s < 0) ? -s : ((s >= NSAMP) ? (2 * NSAMP - 2 - s) : s);
        v = arow[s];
      }
      a[e] = (_Float16)v;
    }
    aF[kc] = a;
  }

  // ---- matmul1: DFT re/im, then power -> LDS in matmul2 A-fragment layout ----
  const v16h* Bre = (const v16h*)BreSw;
  const v16h* Bim = (const v16h*)BimSw;
  for (int nc = 0; nc < 14; ++nc) {
    v8f accR = {};
    v8f accI = {};
    #pragma unroll
    for (int kc = 0; kc < 13; ++kc) {
      v16h br = Bre[(kc * 14 + nc) * 32 + lane];
      v16h bi = Bim[(kc * 14 + nc) * 32 + lane];
      accR = wmma16(aF[kc], br, accR);
      accI = wmma16(aF[kc], bi, accI);
    }
    int f   = nc * 16 + lm;        // freq column this lane owns in D tile
    int kc2 = f >> 5;
    int ff  = f & 31;
    int half_t = (ff >> 3) & 1;    // invert kmap: target half / element
    int et = (((ff >> 4) & 1) << 3) | (((ff >> 1) & 3) << 1) | (ff & 1);
    #pragma unroll
    for (int v = 0; v < 8; ++v) {
      float p = accR[v] * accR[v] + accI[v] * accI[v];
      int mrow = v + (lh << 3);    // D layout: row m = vgpr + 8*half
      powLDS[wave][kc2][mrow + (half_t << 4)][et] = (_Float16)p;
    }
  }

  // ---- matmul2: power (16 x 224) x melFB (224 x 128) -> mel f32 in LDS ----
  const v16h* FB = (const v16h*)FBSw;
  for (int n2 = 0; n2 < 8; ++n2) {
    v8f acc = {};
    #pragma unroll
    for (int kc2 = 0; kc2 < 7; ++kc2) {
      v16h a2 = *(const v16h*)&powLDS[wave][kc2][lane][0];
      v16h bf = FB[(kc2 * 8 + n2) * 32 + lane];
      acc = wmma16(a2, bf, acc);
    }
    #pragma unroll
    for (int v = 0; v < 8; ++v) {
      int tt = wave * 16 + v + (lh << 3);
      if (tt < TFRAMES) melLDS[tt][n2 * 16 + lm] = acc[v];
    }
  }

  __syncthreads();

  // ---- PCEN: sequential IIR scan over T per mel channel, in-place in LDS ----
  if (tid < NMELS) {
    const int j = tid;
    const float sqrt2 = 1.41421356237309515f;
    float ms = melLDS[0][j];
    for (int tt = 0; tt < TFRAMES; ++tt) {
      float e = melLDS[tt][j];
      if (tt > 0) ms = 0.96f * ms + 0.04f * e;
      float denom = powf(ms + 1e-8f, 0.8f);
      melLDS[tt][j] = sqrtf(e / denom + 2.0f) - sqrt2;
    }
  }

  __syncthreads();

  // ---- antialiased linear resize 201 -> 192 (mel axis identity), write f32 out ----
  float* orow = out + (size_t)b * (192 * 128);
  const float scale = 201.0f / 192.0f;     // inv_scale > 1 => widen triangle kernel
  for (int idx = tid; idx < 192 * 128; idx += BLOCKT) {
    int i = idx >> 7;
    int j = idx & 127;
    float x = ((float)i + 0.5f) * scale - 0.5f;
    int t0 = (int)ceilf(x - scale);
    int t1 = (int)floorf(x + scale);
    float sum = 0.0f, wsum = 0.0f;
    for (int tt = t0; tt <= t1; ++tt) {
      if (tt < 0 || tt > TFRAMES - 1) continue;
      float w = 1.0f - fabsf((float)tt - x) / scale;
      if (w > 0.0f) { sum += w * melLDS[tt][j]; wsum += w; }
    }
    orow[idx] = sum / wsum;
  }
}

extern "C" void kernel_launch(void* const* d_in, const int* in_sizes, int n_in,
                              void* d_out, int out_size, void* d_ws, size_t ws_size,
                              hipStream_t stream) {
  (void)in_sizes; (void)n_in; (void)out_size; (void)ws_size;
  const float* audio = (const float*)d_in[0];
  float* out = (float*)d_out;

  _Float16* BreSw = (_Float16*)d_ws;            // 93184 f16
  _Float16* BimSw = BreSw + NB_ELE;             // 93184 f16
  _Float16* FBSw  = BimSw + NB_ELE;             // 28672 f16  (total ~430 KB)

  const int setup_total = 2 * NB_ELE + FB_ELE;  // 215040
  mel_setup_kernel<<<(setup_total + 255) / 256, 256, 0, stream>>>(BreSw, BimSw, FBSw);

  mel_pcen_fused_kernel<<<BATCH, BLOCKT, 0, stream>>>(audio, BreSw, BimSw, FBSw, out);
}
